// STGCNBlock_18880676233328
// MI455X (gfx1250) — matrix operand order repaired and checked by reading, not compiled, not run on hardware
//
#include <hip/hip_runtime.h>

// STGCN block for MI455X (gfx1250, wave32, WMMA).
// GEMM parts use v_wmma_f32_16x16x32_bf16 (f32 accumulate).
// Layout assumptions per CDNA5 ISA:
//   A (16x32 bf16): lanes 0-15 M=lane, K={0..7,16..23}; lanes16-31 M=lane-16, K={8..15,24..31}
//   B (32x16 bf16): VGPR v: lanes0-15 K=2v,2v+1 (lo/hi), lanes16-31 K=2v+8,2v+9; both +8 for v>=4; N=lane%16
//   C/D (16x16 f32): VGPR r: lanes0-15 M=r, lanes16-31 M=r+8; N=lane%16

typedef __attribute__((ext_vector_type(16))) __bf16 v16bf;
typedef __attribute__((ext_vector_type(8)))  float  v8f;

#define B_CT 4
#define N_CT 1024
#define E_CT 16384
#define C_CT 32

static __device__ __forceinline__ v16bf load_arow(const float* __restrict__ r, int ciB) {
  float4 u0 = *(const float4*)(r + ciB);
  float4 u1 = *(const float4*)(r + ciB + 4);
  float4 u2 = *(const float4*)(r + ciB + 16);
  float4 u3 = *(const float4*)(r + ciB + 20);
  v16bf a;
  a[0]  = (__bf16)u0.x; a[1]  = (__bf16)u0.y; a[2]  = (__bf16)u0.z; a[3]  = (__bf16)u0.w;
  a[4]  = (__bf16)u1.x; a[5]  = (__bf16)u1.y; a[6]  = (__bf16)u1.z; a[7]  = (__bf16)u1.w;
  a[8]  = (__bf16)u2.x; a[9]  = (__bf16)u2.y; a[10] = (__bf16)u2.z; a[11] = (__bf16)u2.w;
  a[12] = (__bf16)u3.x; a[13] = (__bf16)u3.y; a[14] = (__bf16)u3.z; a[15] = (__bf16)u3.w;
  return a;
}

// Fast sigmoid: v_exp_f32 + v_rcp_f32 (avoids IEEE divide expansion).
static __device__ __forceinline__ float fast_sigmoid(float q) {
  return __builtin_amdgcn_rcpf(1.f + __expf(-q));
}

// Repack weights into per-lane B fragments (bf16x2 per uint).
// mode 0: conv weight [32co][32ci][3k] -> K-tile kt = tap k, K-within = ci
// mode 1: cheb W [2][32ci][32co]       -> K-tile kt = cheb order
__global__ void pack_w_kernel(const float* __restrict__ w, unsigned* __restrict__ pw,
                              int nkt, int mode) {
  int idx = blockIdx.x * blockDim.x + threadIdx.x;
  if (idx >= nkt * 512) return;
  int v   = idx & 7;
  int lane = (idx >> 3) & 31;
  int n0t = (idx >> 8) & 1;
  int kt  = idx >> 9;
  int Lm  = lane & 15;
  int K0  = 2 * v + ((lane < 16) ? 0 : 8) + ((v >= 4) ? 8 : 0);
  int co  = n0t * 16 + Lm;
  float f0, f1;
  if (mode == 0) { f0 = w[co * 96 + K0 * 3 + kt];       f1 = w[co * 96 + (K0 + 1) * 3 + kt]; }
  else           { f0 = w[kt * 1024 + K0 * 32 + co];    f1 = w[kt * 1024 + (K0 + 1) * 32 + co]; }
  union { __bf16 h; unsigned short u; } a, b;
  a.h = (__bf16)f0; b.h = (__bf16)f1;
  pw[idx] = ((unsigned)b.u << 16) | (unsigned)a.u;
}

// Gated temporal conv: one wave per (b,n). GEMM M=64(pad),K=96,N=32 via WMMA bf16.
// B fragments (9 x 8 VGPRs) hoisted across the M loop; A rebuilt from LDS per tile.
__global__ __launch_bounds__(32)
void tconv_kernel(const float* __restrict__ in, float* __restrict__ out,
                  const unsigned* __restrict__ pwP, const unsigned* __restrict__ pwQ,
                  const unsigned* __restrict__ pwR,
                  const float* __restrict__ bP, const float* __restrict__ bQ,
                  const float* __restrict__ bR, int Tin, int Tout) {
  __shared__ float xs[64][32];
  int b = blockIdx.x >> 10;
  int n = blockIdx.x & (N_CT - 1);
  int lane = threadIdx.x;

  const float* src = in + ((long)(b * Tin) * N_CT + n) * C_CT;
  // Warm L2/L0 for the neighbouring tile (gfx1250 global_prefetch_b8).
  __builtin_prefetch(src + N_CT * C_CT, 0, 1);

  for (int idx = lane; idx < 64 * 8; idx += 32) {
    int row = idx >> 3, q = (idx & 7) * 4;
    float4 val = make_float4(0.f, 0.f, 0.f, 0.f);
    if (row < Tin) val = *(const float4*)(src + (long)row * N_CT * C_CT + q);
    *(float4*)(&xs[row][q]) = val;
  }
  __syncthreads();

  int Lm  = lane & 15;
  int ciB = (lane < 16) ? 0 : 8;

  for (int n0t = 0; n0t < 2; ++n0t) {
    // Hoisted B fragments: invariant over the M loop.
    v16bf fP[3], fQ[3], fR[3];
#pragma unroll
    for (int kt = 0; kt < 3; ++kt) {
      int base = ((kt * 2 + n0t) * 32 + lane) * 8;
      fP[kt] = *(const v16bf*)(pwP + base);
      fQ[kt] = *(const v16bf*)(pwQ + base);
      fR[kt] = *(const v16bf*)(pwR + base);
    }
    int co = n0t * 16 + Lm;
    float biP = bP[co], biQ = bQ[co], biR = bR[co];

    for (int m = 0; m < 4; ++m) {
      v16bf afr[3];
#pragma unroll
      for (int kt = 0; kt < 3; ++kt) {
        int t = m * 16 + Lm + kt;
        if (t < 64) afr[kt] = load_arow(&xs[t][0], ciB);
        else {
          v16bf z;
#pragma unroll
          for (int i = 0; i < 16; ++i) z[i] = (__bf16)0.f;
          afr[kt] = z;
        }
      }
      v8f accP = {}, accQ = {}, accR = {};
#pragma unroll
      for (int kt = 0; kt < 3; ++kt) {
        accP = __builtin_amdgcn_wmma_f32_16x16x32_bf16(false, afr[kt], false, fP[kt], (short)0, accP, false, false);
        accQ = __builtin_amdgcn_wmma_f32_16x16x32_bf16(false, afr[kt], false, fQ[kt], (short)0, accQ, false, false);
        accR = __builtin_amdgcn_wmma_f32_16x16x32_bf16(false, afr[kt], false, fR[kt], (short)0, accR, false, false);
      }
#pragma unroll
      for (int r = 0; r < 8; ++r) {
        int tp = m * 16 + ((lane < 16) ? r : (r + 8));
        if (tp < Tout) {
          float p  = accP[r] + biP;
          float q  = accQ[r] + biQ;
          float rr = accR[r] + biR;
          float hv = p * fast_sigmoid(q) + rr;
          out[(((long)(b * Tout + tp)) * N_CT + n) * C_CT + co] = fmaxf(hv, 0.f);
        }
      }
    }
  }
}

__global__ void fill0_kernel(float* __restrict__ p, long n) {
  long i = (long)blockIdx.x * blockDim.x + threadIdx.x;
  long stride = (long)gridDim.x * blockDim.x;
  for (; i < n; i += stride) p[i] = 0.f;
}

__global__ void deg_kernel(const long long* __restrict__ ei, const float* __restrict__ ew,
                           float* __restrict__ deg) {
  int i = blockIdx.x * blockDim.x + threadIdx.x;
  if (i >= E_CT) return;
  int s = (int)ei[i], d = (int)ei[E_CT + i];
  float w = (s == d) ? 0.f : ew[i];
  atomicAdd(&deg[s], w);
}

__global__ void norm_kernel(const long long* __restrict__ ei, const float* __restrict__ ew,
                            const float* __restrict__ deg, float* __restrict__ nrm) {
  int i = blockIdx.x * blockDim.x + threadIdx.x;
  if (i >= E_CT) return;
  int s = (int)ei[i], d = (int)ei[E_CT + i];
  float w  = (s == d) ? 0.f : ew[i];
  float ds = deg[s], dd = deg[d];
  float is = ds > 0.f ? rsqrtf(ds) : 0.f;
  float id = dd > 0.f ? rsqrtf(dd) : 0.f;
  nrm[i] = -is * w * id;
}

// One wave per (edge, b*t'): lane = channel; L2-resident f32 atomics.
__global__ __launch_bounds__(256)
void scatter_kernel(const float* __restrict__ h, const long long* __restrict__ ei,
                    const float* __restrict__ nrm, float* __restrict__ Lh) {
  int e    = blockIdx.x * 8 + (threadIdx.x >> 5);
  int bt   = blockIdx.y;
  int lane = threadIdx.x & 31;
  long s = ei[e], d = ei[E_CT + e];
  float w = nrm[e];
  float v = h[((long)bt * N_CT + s) * C_CT + lane] * w;
  atomicAdd(&Lh[((long)bt * N_CT + d) * C_CT + lane], v);
}

// out = relu([h|Lh] x [W0;W1] + bias). One wave per 16 rows.
__global__ __launch_bounds__(256)
void cheb_kernel(const float* __restrict__ h, const float* __restrict__ Lh,
                 const unsigned* __restrict__ pwC, const float* __restrict__ bias,
                 float* __restrict__ out, int Mrows) {
  int wid  = blockIdx.x * 8 + (threadIdx.x >> 5);
  int lane = threadIdx.x & 31;
  int m0 = wid * 16;
  if (m0 >= Mrows) return;
  int Lm  = lane & 15;
  int ciB = (lane < 16) ? 0 : 8;
  int row = m0 + Lm;
  __builtin_prefetch(h + (long)(row + 16) * 32, 0, 1);
  v16bf a0 = load_arow(h  + (long)row * 32, ciB);
  v16bf a1 = load_arow(Lh + (long)row * 32, ciB);
#pragma unroll
  for (int n0t = 0; n0t < 2; ++n0t) {
    v8f acc = {};
    v16bf b0 = *(const v16bf*)(pwC + ((0 * 2 + n0t) * 32 + lane) * 8);
    v16bf b1 = *(const v16bf*)(pwC + ((1 * 2 + n0t) * 32 + lane) * 8);
    acc = __builtin_amdgcn_wmma_f32_16x16x32_bf16(false, a0, false, b0, (short)0, acc, false, false);
    acc = __builtin_amdgcn_wmma_f32_16x16x32_bf16(false, a1, false, b1, (short)0, acc, false, false);
    int co = n0t * 16 + Lm;
    float bs = bias[co];
#pragma unroll
    for (int r = 0; r < 8; ++r) {
      int mr = m0 + ((lane < 16) ? r : (r + 8));
      out[(long)mr * 32 + co] = fmaxf(acc[r] + bs, 0.f);
    }
  }
}

// BatchNorm over (B,T'',C) per node; one workgroup per node.
__global__ __launch_bounds__(256)
void bn_kernel(const float* __restrict__ h, const float* __restrict__ gamma,
               const float* __restrict__ beta, float* __restrict__ out) {
  __shared__ float s1[256], s2[256];
  int node = blockIdx.x;
  int tid  = threadIdx.x;
  const int TOT = B_CT * 60 * C_CT; // 7680
  float s = 0.f, sq = 0.f;
  for (int j = tid; j < TOT; j += 256) {
    int bt = j >> 5, c = j & 31;
    float v = h[((long)bt * N_CT + node) * C_CT + c];
    s += v; sq += v * v;
  }
  s1[tid] = s; s2[tid] = sq;
  __syncthreads();
  for (int off = 128; off > 0; off >>= 1) {
    if (tid < off) { s1[tid] += s1[tid + off]; s2[tid] += s2[tid + off]; }
    __syncthreads();
  }
  float mean = s1[0] / (float)TOT;
  float var  = s2[0] / (float)TOT - mean * mean;
  float sc = rsqrtf(var + 1e-5f) * gamma[node];
  float sh = beta[node] - mean * sc;
  for (int j = tid; j < TOT; j += 256) {
    int bt = j >> 5, c = j & 31;
    long idx = ((long)bt * N_CT + node) * C_CT + c;
    out[idx] = h[idx] * sc + sh;
  }
}

extern "C" void kernel_launch(void* const* d_in, const int* in_sizes, int n_in,
                              void* d_out, int out_size, void* d_ws, size_t ws_size,
                              hipStream_t stream) {
  const float*     x   = (const float*)d_in[0];
  const long long* ei  = (const long long*)d_in[1];
  const float*     ew  = (const float*)d_in[2];
  const float* w11 = (const float*)d_in[3];
  const float* b11 = (const float*)d_in[4];
  const float* w12 = (const float*)d_in[5];
  const float* b12 = (const float*)d_in[6];
  const float* w13 = (const float*)d_in[7];
  const float* b13 = (const float*)d_in[8];
  const float* chW = (const float*)d_in[9];
  const float* chb = (const float*)d_in[10];
  const float* w21 = (const float*)d_in[11];
  const float* b21 = (const float*)d_in[12];
  const float* w22 = (const float*)d_in[13];
  const float* b22 = (const float*)d_in[14];
  const float* w23 = (const float*)d_in[15];
  const float* b23 = (const float*)d_in[16];
  const float* gam = (const float*)d_in[17];
  const float* bet = (const float*)d_in[18];

  const long H1S = (long)B_CT * 62 * N_CT * C_CT; // 8,126,464 floats
  char* ws = (char*)d_ws;
  float*    H1  = (float*)(ws);               // [B,62,N,32]
  float*    H2  = (float*)(ws + 32505856);    // [B,62,N,32]
  float*    LH3 = (float*)(ws + 65011712);    // Lh, later reused as H3 [B,60,N,32]
  float*    deg = (float*)(ws + 97517568);
  float*    nrm = deg + 1024;
  unsigned* pw  = (unsigned*)(nrm + 16384);
  unsigned *pw0 = pw,        *pw1 = pw + 1536, *pw2 = pw + 3072,
           *pw3 = pw + 4608, *pw4 = pw + 6144, *pw5 = pw + 7680,
           *pwC = pw + 9216;

  // Repack weights into WMMA B fragments (bf16)
  pack_w_kernel<<<6, 256, 0, stream>>>(w11, pw0, 3, 0);
  pack_w_kernel<<<6, 256, 0, stream>>>(w12, pw1, 3, 0);
  pack_w_kernel<<<6, 256, 0, stream>>>(w13, pw2, 3, 0);
  pack_w_kernel<<<6, 256, 0, stream>>>(w21, pw3, 3, 0);
  pack_w_kernel<<<6, 256, 0, stream>>>(w22, pw4, 3, 0);
  pack_w_kernel<<<6, 256, 0, stream>>>(w23, pw5, 3, 0);
  pack_w_kernel<<<4, 256, 0, stream>>>(chW, pwC, 2, 1);

  fill0_kernel<<<4, 256, 0, stream>>>(deg, 1024);
  fill0_kernel<<<4096, 256, 0, stream>>>(LH3, H1S);

  deg_kernel<<<64, 256, 0, stream>>>(ei, ew, deg);
  norm_kernel<<<64, 256, 0, stream>>>(ei, ew, deg, nrm);

  // temporal conv 1: x[B,64,N,32] -> H1[B,62,N,32]
  tconv_kernel<<<B_CT * N_CT, 32, 0, stream>>>(x, H1, pw0, pw1, pw2, b11, b12, b13, 64, 62);

  // edge scatter: Lh += norm * h[src]
  scatter_kernel<<<dim3(E_CT / 8, B_CT * 62), 256, 0, stream>>>(H1, ei, nrm, LH3);

  // cheb GEMM + relu: H2 = relu([H1|Lh] x [W0;W1] + b)
  int Mrows = B_CT * 62 * N_CT;             // 253,952
  cheb_kernel<<<Mrows / 16 / 8, 256, 0, stream>>>(H1, LH3, pwC, chb, H2, Mrows);

  // temporal conv 2: H2[B,62,N,32] -> H3[B,60,N,32] (reuses Lh buffer)
  tconv_kernel<<<B_CT * N_CT, 32, 0, stream>>>(H2, LH3, pw3, pw4, pw5, b21, b22, b23, 62, 60);

  // per-node batch norm -> d_out
  bn_kernel<<<N_CT, 256, 0, stream>>>(LH3, gam, bet, (float*)d_out);
}
